// SparseAttention_38319698215041
// MI455X (gfx1250) — matrix operand order repaired and checked by reading, not compiled, not run on hardware
//
#include <hip/hip_runtime.h>
#include <hip/hip_bf16.h>

typedef __bf16 bf16;
typedef __attribute__((ext_vector_type(16))) __bf16 v16bf;
typedef __attribute__((ext_vector_type(8)))  __bf16 v8bf;
typedef __attribute__((ext_vector_type(8)))  float  v8f;
typedef __attribute__((ext_vector_type(4)))  unsigned int v4u;
typedef __attribute__((ext_vector_type(8)))  int v8i;
typedef __attribute__((ext_vector_type(4)))  int v4i;

#define BB    2
#define SS    2048
#define HIDD  1024
#define HH    16
#define DD    64
#define TT    (BB * SS)
#define KKEEP 204   // int(2048 * (1 - 0.9))

// LDS layout of the attention kernel (byte offsets into dynamic shared mem)
#define OFF_SC   0                       // [16][SS]   f32 scores   (128 KB)
#define OFF_PW   131072                  // [16][SS]   bf16 probs   ( 64 KB)
#define OFF_RED  196608                  // [4][16][16] f32 reduce  (  4 KB)
#define OFF_K0   200704                  // K stage buf 0 (128 rows * 144 B)
#define OFF_K1   219136                  // K stage buf 1
#define SMEM_SZ  237568
#define KROWPAD  72                      // LDS K row stride in bf16 (144 B)

// ---------------------------------------------------------------------------
// Fragment loaders matching CDNA5 WMMA 16-bit layouts (CDNA5 ISA)
// ---------------------------------------------------------------------------
static __device__ inline v16bf load_a16(const bf16* p) {
  v8bf lo = *(const v8bf*)p;
  v8bf hi = *(const v8bf*)(p + 16);
  v16bf r;
#pragma unroll
  for (int i = 0; i < 8; ++i) { r[i] = lo[i]; r[8 + i] = hi[i]; }
  return r;
}
static __device__ inline v16bf load_b16(const bf16* p) {
  v8bf lo = *(const v8bf*)p;
  v8bf hi = *(const v8bf*)(p + 16);
  v16bf r;
#pragma unroll
  for (int i = 0; i < 8; ++i) { r[i] = lo[i]; r[8 + i] = hi[i]; }
  return r;
}

// ---------------------------------------------------------------------------
// TDM: stage a 128-row x 64-col bf16 K slab (row stride HID) into LDS.
// D# group0: count=1, lds_addr, 57-bit global addr, type=2.
// D# group1: data_size=2B, pad_enable (32-dword interval, 4-dword pad ->
//            144 B LDS rows, 16B aligned), tensor dims/strides, 2D tile.
// ---------------------------------------------------------------------------
static __device__ inline void tdm_load_k(unsigned long long gaddr,
                                         unsigned lds_off) {
  v4u g0;
  g0[0] = 1u;                                        // count=1 (valid user D#)
  g0[1] = lds_off;                                   // LDS byte address
  g0[2] = (unsigned)(gaddr & 0xFFFFFFFFu);           // global_addr[31:0]
  g0[3] = (unsigned)((gaddr >> 32) & 0x1FFFFFFu)     // global_addr[56:32]
        | (2u << 30);                                // type = 2 ("image")
  v8i g1;
  g1[0] = (1 << 16)                                  // data_size = 2 bytes
        | (1 << 20)                                  // pad_enable
        | (4 << 22)                                  // pad_interval = 32 dwords
        | (3 << 25);                                 // pad_amount   =  4 dwords
  g1[1] = (int)((HIDD & 0xFFFF) << 16);              // tensor_dim0[15:0]
  g1[2] = (int)((HIDD >> 16) | ((SS & 0xFFFF) << 16)); // dim0 hi | dim1 lo
  g1[3] = (int)((SS >> 16) | (DD << 16));            // dim1 hi | tile_dim0=64
  g1[4] = 128;                                       // tile_dim1=128, dim2=0
  g1[5] = HIDD;                                      // tensor_dim0_stride lo32
  g1[6] = 0;                                         // stride0 hi | stride1 lo
  g1[7] = 0;
  v4i z = {};
#if __clang_major__ >= 23
  v8i z8 = {};
  __builtin_amdgcn_tensor_load_to_lds(g0, g1, z, z, z8, 0);
#else
  __builtin_amdgcn_tensor_load_to_lds(g0, g1, z, z, 0);
#endif
}

// ---------------------------------------------------------------------------
// fp32 -> bf16 conversion / weight transpose (Wt[n][k] = W[k][n])
// ---------------------------------------------------------------------------
__global__ void cvt_bf16_kernel(const float* __restrict__ in,
                                bf16* __restrict__ out, int n) {
  int i = blockIdx.x * blockDim.x + threadIdx.x;
  if (i < n) out[i] = (bf16)in[i];
}

__global__ void transpose_w_kernel(const float* __restrict__ W,
                                   bf16* __restrict__ Wt) {
  int i = blockIdx.x * blockDim.x + threadIdx.x;   // over HID*HID
  int k = i >> 10, n = i & (HIDD - 1);
  Wt[(size_t)n * HIDD + k] = (bf16)W[i];
}

// ---------------------------------------------------------------------------
// GEMM: Y[t][n] = A[t][:] @ Wt[n][:] + bias[n]
// Wave tile 32x64 (2 A-frags x 4 B-frags -> 8 WMMA / 12 b128 loads,
// ~21 flop/byte from cache). Block = 8 waves -> 64x256 tile.
// mode 0: out bf16 [TT,HID]; mode 1: V^T [B][H][D][S] bf16; mode 2: f32.
// ---------------------------------------------------------------------------
__global__ __launch_bounds__(256)
void gemm_bf16_kernel(const bf16* __restrict__ A,
                      const bf16* __restrict__ Bt,
                      const float* __restrict__ bias,
                      void* __restrict__ Out, int mode) {
  const int lane = threadIdx.x & 31;
  const int w    = threadIdx.x >> 5;
  const int m0   = blockIdx.x * 64  + (w & 1) * 32;   // 32 rows per wave
  const int nb   = blockIdx.y * 256 + (w >> 1) * 64;  // 64 cols per wave
  const int asub = lane >> 4;

  const bf16* ap0 = A + (size_t)(m0 + (lane & 15)) * HIDD + asub * 8;
  const bf16* ap1 = ap0 + (size_t)16 * HIDD;
  const bf16* bp  = Bt + (size_t)(nb + (lane & 15)) * HIDD + asub * 8;

  v8f acc[2][4] = {};
  for (int k0 = 0; k0 < HIDD; k0 += 32) {
    __builtin_prefetch(ap0 + k0 + 128, 0, 1);         // global_prefetch_b8
    v16bf a0 = load_a16(ap0 + k0);
    v16bf a1 = load_a16(ap1 + k0);
    v16bf bf[4];
#pragma unroll
    for (int j = 0; j < 4; ++j)
      bf[j] = load_b16(bp + (size_t)j * 16 * HIDD + k0);
#pragma unroll
    for (int j = 0; j < 4; ++j) {
      acc[0][j] = __builtin_amdgcn_wmma_f32_16x16x32_bf16(false, a0, false, bf[j],
                                                          (short)0, acc[0][j],
                                                          false, false);
      acc[1][j] = __builtin_amdgcn_wmma_f32_16x16x32_bf16(false, a1, false, bf[j],
                                                          (short)0, acc[1][j],
                                                          false, false);
    }
  }

#pragma unroll
  for (int mi = 0; mi < 2; ++mi) {
#pragma unroll
    for (int j = 0; j < 4; ++j) {
      const int ncol = nb + j * 16 + (lane & 15);
      const float bv = bias[ncol];
#pragma unroll
      for (int r = 0; r < 8; ++r) {
        const int row = m0 + mi * 16 + asub * 8 + r;
        const float v = acc[mi][j][r] + bv;
        if (mode == 0) {
          ((bf16*)Out)[(size_t)row * HIDD + ncol] = (bf16)v;
        } else if (mode == 1) {                     // V -> V^T [B][H][D][S]
          const int bidx = row / SS, sidx = row % SS;
          const int h = ncol / DD, dd = ncol % DD;
          ((bf16*)Out)[(((size_t)bidx * HH + h) * DD + dd) * SS + sidx] = (bf16)v;
        } else {
          ((float*)Out)[(size_t)row * HIDD + ncol] = v;
        }
      }
    }
  }
}

// ---------------------------------------------------------------------------
// Attention: one block per (b, h, 16-query tile). 256 thr = 8 waves.
// Phase 1: K slabs staged to LDS by the Tensor Data Mover, double-buffered
//          with s_wait_tensorcnt; scores via WMMA into LDS (f32).
// Phase 2: per-row top-204 threshold (wave32 bisection) + softmax weights.
// Phase 3: dense probs @ V^T (WMMA), K range split across wave halves.
// ---------------------------------------------------------------------------
__global__ __launch_bounds__(256)
void attn_kernel(const bf16* __restrict__ Qb, const bf16* __restrict__ Kb,
                 const bf16* __restrict__ Vt, bf16* __restrict__ OutH) {
  extern __shared__ char smem[];
  float* sc  = (float*)(smem + OFF_SC);
  bf16*  pw  = (bf16*)(smem + OFF_PW);
  float* red = (float*)(smem + OFF_RED);

  const int lane = threadIdx.x & 31;
  const int w    = threadIdx.x >> 5;
  const int qt = blockIdx.x & 127;          // S/16 = 128
  const int h  = (blockIdx.x >> 7) & 15;
  const int b  = blockIdx.x >> 11;
  const int q0 = qt * 16;

  // ---- Phase 1: scores = (Q K^T) / sqrt(64) -> LDS ----
  const bf16* qp = Qb + (size_t)(b * SS + q0 + (lane & 15)) * HIDD
                      + h * DD + (lane >> 4) * 8;
  const v16bf a0 = load_a16(qp);        // K dims  0..31
  const v16bf a1 = load_a16(qp + 32);   // K dims 32..63
  const unsigned long long kbase =
      (unsigned long long)(const void*)(Kb + (size_t)(b * SS) * HIDD + h * DD);

  if (w == 0) tdm_load_k(kbase, OFF_K0);            // prologue: slab 0
  for (int blk = 0; blk < SS / 128; ++blk) {
    if (w == 0) {
      if (blk + 1 < SS / 128) {                     // keep next slab in flight
        tdm_load_k(kbase + (unsigned long long)(blk + 1) * 128 * HIDD * 2,
                   ((blk + 1) & 1) ? OFF_K1 : OFF_K0);
        __builtin_amdgcn_s_wait_tensorcnt(1);       // slab `blk` landed
      } else {
        __builtin_amdgcn_s_wait_tensorcnt(0);       // drain final slab
      }
    }
    __syncthreads();                                // slab `blk` visible in LDS
    const bf16* kls = (const bf16*)(smem + ((blk & 1) ? OFF_K1 : OFF_K0));
    const bf16* kp  = kls + (size_t)(w * 16 + (lane & 15)) * KROWPAD
                          + (lane >> 4) * 8;
    v16bf b0 = load_b16(kp);                        // ds_load_b128 x2
    v16bf b1 = load_b16(kp + 32);
    v8f c = {};
    c = __builtin_amdgcn_wmma_f32_16x16x32_bf16(false, a0, false, b0,
                                                (short)0, c, false, false);
    c = __builtin_amdgcn_wmma_f32_16x16x32_bf16(false, a1, false, b1,
                                                (short)0, c, false, false);
    const int n0 = blk * 128 + w * 16;
#pragma unroll
    for (int r = 0; r < 8; ++r)
      sc[((lane >> 4) * 8 + r) * SS + n0 + (lane & 15)] = c[r] * 0.125f;
    __syncthreads();                                // done reading slab buffer
  }

  // ---- Phase 2: per-row top-204 threshold + softmax weights ----
  for (int rr = 0; rr < 2; ++rr) {
    const int r = w * 2 + rr;
    const float* row = sc + (size_t)r * SS;
    float vmax = -1e30f, vmin = 1e30f;
    for (int j = lane; j < SS; j += 32) {
      const float v = row[j];
      vmax = fmaxf(vmax, v); vmin = fminf(vmin, v);
    }
#pragma unroll
    for (int m = 16; m >= 1; m >>= 1) {
      vmax = fmaxf(vmax, __shfl_xor(vmax, m, 32));
      vmin = fminf(vmin, __shfl_xor(vmin, m, 32));
    }
    float lo = vmin, hi = vmax;
    for (int it = 0; it < 22; ++it) {            // value-space bisection
      const float t = 0.5f * (lo + hi);
      int cnt = 0;
      for (int j = lane; j < SS; j += 32) cnt += (row[j] > t) ? 1 : 0;
#pragma unroll
      for (int m = 16; m >= 1; m >>= 1) cnt += __shfl_xor(cnt, m, 32);
      if (cnt > KKEEP) lo = t; else hi = t;
    }
    const float t = lo;
    const float mval = fmaxf(vmax, 0.0f);        // zeros participate in softmax
    float se = 0.0f; int cnt = 0;
    for (int j = lane; j < SS; j += 32) {
      const float v = row[j];
      if (v > t) { se += __expf(v - mval); cnt++; }
    }
#pragma unroll
    for (int m = 16; m >= 1; m >>= 1) {
      se  += __shfl_xor(se, m, 32);
      cnt += __shfl_xor(cnt, m, 32);
    }
    const float e0  = __expf(-mval);
    const float inv = 1.0f / (se + (float)(SS - cnt) * e0);
    const float w0  = e0 * inv;                  // floor weight of zeros
    for (int j = lane; j < SS; j += 32) {
      const float v = row[j];
      const float p = (v > t) ? __expf(v - mval) * inv : w0;
      pw[(size_t)r * SS + j] = (bf16)p;
    }
  }
  __syncthreads();

  // ---- Phase 3: out = probs @ V  (dense; B from V^T so K is contiguous) ----
  const int nt    = w & 3;                       // 16-dim slab of D=64
  const int khalf = w >> 2;                      // split K range over wave halves
  const bf16* vp = Vt + ((size_t)(b * HH + h) * DD + nt * 16 + (lane & 15)) * SS
                      + (lane >> 4) * 8;
  const bf16* awp = pw + (size_t)(lane & 15) * SS + (lane >> 4) * 8;
  v8f c = {};
  for (int k0 = khalf * 1024; k0 < khalf * 1024 + 1024; k0 += 32) {
    v16bf a  = load_a16(awp + k0);
    v16bf bb = load_b16(vp + k0);
    c = __builtin_amdgcn_wmma_f32_16x16x32_bf16(false, a, false, bb,
                                                (short)0, c, false, false);
  }
  if (khalf == 1) {
#pragma unroll
    for (int r = 0; r < 8; ++r)
      red[(nt * 16 + (lane >> 4) * 8 + r) * 16 + (lane & 15)] = c[r];
  }
  __syncthreads();
  if (khalf == 0) {
#pragma unroll
    for (int r = 0; r < 8; ++r) {
      const float v = c[r] + red[(nt * 16 + (lane >> 4) * 8 + r) * 16 + (lane & 15)];
      const int token = b * SS + q0 + (lane >> 4) * 8 + r;
      const int col   = h * DD + nt * 16 + (lane & 15);
      OutH[(size_t)token * HIDD + col] = (bf16)v;   // heads re-merged in place
    }
  }
}

// ---------------------------------------------------------------------------
extern "C" void kernel_launch(void* const* d_in, const int* in_sizes, int n_in,
                              void* d_out, int out_size, void* d_ws, size_t ws_size,
                              hipStream_t stream) {
  (void)in_sizes; (void)n_in; (void)out_size; (void)ws_size;
  const float* x  = (const float*)d_in[0];
  const float* Wq = (const float*)d_in[1];
  const float* bq = (const float*)d_in[2];
  const float* Wk = (const float*)d_in[3];
  const float* bk = (const float*)d_in[4];
  const float* Wv = (const float*)d_in[5];
  const float* bv = (const float*)d_in[6];
  const float* Wo = (const float*)d_in[7];
  const float* bo = (const float*)d_in[8];

  char* ws = (char*)d_ws;
  bf16* xb   = (bf16*)(ws);                     //  8 MB  x in bf16
  bf16* Wqt  = (bf16*)(ws + ( 8ull << 20));     //  2 MB  each, transposed
  bf16* Wkt  = (bf16*)(ws + (10ull << 20));
  bf16* Wvt  = (bf16*)(ws + (12ull << 20));
  bf16* Wot  = (bf16*)(ws + (14ull << 20));
  bf16* Qb   = (bf16*)(ws + (16ull << 20));     //  8 MB  [B,S,HID]
  bf16* Kb   = (bf16*)(ws + (24ull << 20));     //  8 MB  [B,S,HID]
  bf16* Vt   = (bf16*)(ws + (32ull << 20));     //  8 MB  [B,H,D,S]
  bf16* OutH = (bf16*)(ws + (40ull << 20));     //  8 MB  [B,S,HID]

  cvt_bf16_kernel<<<(TT * HIDD) / 256, 256, 0, stream>>>(x, xb, TT * HIDD);
  transpose_w_kernel<<<(HIDD * HIDD) / 256, 256, 0, stream>>>(Wq, Wqt);
  transpose_w_kernel<<<(HIDD * HIDD) / 256, 256, 0, stream>>>(Wk, Wkt);
  transpose_w_kernel<<<(HIDD * HIDD) / 256, 256, 0, stream>>>(Wv, Wvt);
  transpose_w_kernel<<<(HIDD * HIDD) / 256, 256, 0, stream>>>(Wo, Wot);

  dim3 gg(TT / 64, HIDD / 256);
  gemm_bf16_kernel<<<gg, 256, 0, stream>>>(xb, Wqt, bq, (void*)Qb, 0);
  gemm_bf16_kernel<<<gg, 256, 0, stream>>>(xb, Wkt, bk, (void*)Kb, 0);
  gemm_bf16_kernel<<<gg, 256, 0, stream>>>(xb, Wvt, bv, (void*)Vt, 1);

  attn_kernel<<<BB * HH * (SS / 16), 256, SMEM_SZ, stream>>>(Qb, Kb, Vt, OutH);

  gemm_bf16_kernel<<<gg, 256, 0, stream>>>(OutH, Wot, bo, d_out, 2);
}
